// GraphNetworkLayer_23450521436275
// MI455X (gfx1250) — matrix operand order repaired and checked by reading, not compiled, not run on hardware
//
#include <hip/hip_runtime.h>
#include <math.h>

// ---------------------------------------------------------------------------
// GraphNetwork layer for MI455X (gfx1250, wave32, WMMA).
//   edge MLP : [E,512] @ eW1 -> relu -> @ eW2      (K tiles: 16)
//   node MLP : [N,384] @ nW1 -> relu -> @ nW2      (K tiles: 12)
//   glob MLP : [16,384] @ gW1 -> relu -> @ gW2     (K tiles: 12)
// All GEMMs run on v_wmma_f32_16x16x32_bf16; gathers are fused into the
// A-fragment build, scatters (segment_sum) into the epilogue via f32 atomics.
// f32->bf16 uses native v_cvt_pk_bf16_f32; B fragments are batch-loaded per
// K tile so WMMAs issue under staggered loadcnt waits (latency hiding).
// ---------------------------------------------------------------------------

#define NNODES 131072
#define NEDGES 262144
#define NBATCH 16
#define DDIM   128

typedef float  v8f   __attribute__((ext_vector_type(8)));
typedef float  v8ff  __attribute__((ext_vector_type(8)));
typedef __bf16 v16bf __attribute__((ext_vector_type(16)));
typedef __bf16 v8bf  __attribute__((ext_vector_type(8)));

// Native f32 -> bf16 (hardware cvt on gfx1250)
static __device__ __forceinline__ __bf16 f2bf(float f) { return (__bf16)f; }

static __device__ __forceinline__ v8f wmma_bf16(v16bf a, v16bf b, v8f c) {
  // (neg_a, A, neg_b, B, c_mod, C, reuse_a, reuse_b)
  return __builtin_amdgcn_wmma_f32_16x16x32_bf16(false, a, false, b,
                                                 (short)0, c, false, false);
}

// Build this lane's slice of a 16x32 bf16 A fragment from two runs of 8
// consecutive f32 values. Per CDNA5 ISA (16-bit A, wave32):
//   lane = M + 16*khalf ; elems 0..7 -> K = khalf*8 .. +7
//                         elems 8..15 -> K = 16 + khalf*8 .. +7
static __device__ __forceinline__ v16bf make_a8(const float* r0, const float* r1) {
  v8ff lo = *(const v8ff*)(r0);   // 2x global_load_b128
  v8ff hi = *(const v8ff*)(r1);
  v8bf blo = __builtin_convertvector(lo, v8bf);   // v_cvt_pk_bf16_f32 x4
  v8bf bhi = __builtin_convertvector(hi, v8bf);
  return __builtin_shufflevector(blo, bhi, 0, 1, 2, 3, 4, 5, 6, 7,
                                 8, 9, 10, 11, 12, 13, 14, 15);
}

// Layer-1 GEMM over KT 32-wide K tiles. segs[kt>>2] is the concat-segment base
// row pointer for this lane's M row (K tiles never straddle 128-wide segments).
// Weights are pre-packed in B-fragment order: wf[((kt*8+ct)<<5) + lane].
// All 8 B fragments of a K tile are loaded before the WMMA burst so the
// loads pipeline under staggered s_wait_loadcnt thresholds.
template <int KT>
static __device__ __forceinline__ void gemm_layer1(const float* const segs[4],
                                                   const v16bf* wf, int lane,
                                                   v8f acc[8]) {
  const int kh = lane >> 4;
  #pragma unroll
  for (int kt = 0; kt < KT; ++kt) {
    const float* s = segs[kt >> 2];
    const int off = (kt & 3) * 32 + kh * 8;
    v16bf a = make_a8(s + off, s + off + 16);
    v16bf b[8];
    #pragma unroll
    for (int ct = 0; ct < 8; ++ct) b[ct] = wf[((kt * 8 + ct) << 5) + lane];
    #pragma unroll
    for (int ct = 0; ct < 8; ++ct) acc[ct] = wmma_bf16(a, b[ct], acc[ct]);
  }
}

// Bias + ReLU on the 16x128 f32 accumulator (D layout: n=lane&15, m=r+8*kh),
// written to per-wave LDS staging as row-major bf16 [16][128].
static __device__ __forceinline__ void bias_relu_store(const v8f acc[8],
                                                       const float* bias,
                                                       __bf16* hs, int lane) {
  const int nn = lane & 15, kh = lane >> 4;
  #pragma unroll
  for (int ct = 0; ct < 8; ++ct) {
    const float bv = bias[ct * 16 + nn];
    #pragma unroll
    for (int r = 0; r < 8; ++r) {
      float h = acc[ct][r] + bv;
      h = h > 0.f ? h : 0.f;
      hs[(r + 8 * kh) * DDIM + ct * 16 + nn] = f2bf(h);
    }
  }
}

// Layer-2 GEMM: K = 128, A re-read from LDS staging in A-fragment layout
// (two contiguous 16B ds_load per fragment; per-wave LDS is in-order).
static __device__ __forceinline__ void gemm_layer2(const __bf16* hs,
                                                   const v16bf* wf, int lane,
                                                   v8f acc[8]) {
  const int nn = lane & 15, kh = lane >> 4;
  #pragma unroll
  for (int kt = 0; kt < 4; ++kt) {
    const __bf16* hp = hs + nn * DDIM + kt * 32 + kh * 8;
    v8bf lo = *(const v8bf*)(hp);
    v8bf hi = *(const v8bf*)(hp + 16);
    v16bf a = __builtin_shufflevector(lo, hi, 0, 1, 2, 3, 4, 5, 6, 7,
                                      8, 9, 10, 11, 12, 13, 14, 15);
    v16bf b[8];
    #pragma unroll
    for (int ct = 0; ct < 8; ++ct) b[ct] = wf[((kt * 8 + ct) << 5) + lane];
    #pragma unroll
    for (int ct = 0; ct < 8; ++ct) acc[ct] = wmma_bf16(a, b[ct], acc[ct]);
  }
}

// ---------------------------------------------------------------------------
// Prep kernels
// ---------------------------------------------------------------------------

__global__ void zero_f32_kernel(float* __restrict__ p, int n) {
  int i = blockIdx.x * blockDim.x + threadIdx.x;
  int stride = gridDim.x * blockDim.x;
  for (; i < n; i += stride) p[i] = 0.f;
}

// Pack f32 [K,128] weights into bf16 B-fragment order:
// flat = ((kt*8+ct)*32 + lane)*16 + i ; k = kt*32 + (lane>>4)*8 + (i&7) + (i>>3)*16 ;
// c = ct*16 + (lane&15).
__global__ void pack_weights_kernel(const float* __restrict__ W,
                                    __bf16* __restrict__ dst, int K) {
  int tid = blockIdx.x * blockDim.x + threadIdx.x;
  if (tid >= K * DDIM) return;
  int i    = tid & 15;
  int lane = (tid >> 4) & 31;
  int f    = tid >> 9;
  int kt = f >> 3, ct = f & 7;
  int k = kt * 32 + (lane >> 4) * 8 + (i & 7) + (i >> 3) * 16;
  int c = ct * 16 + (lane & 15);
  dst[tid] = f2bf(W[k * DDIM + c]);
}

// ---------------------------------------------------------------------------
// Edge block: u = relu([edges|sent|recv|g] @ eW1 + b1) @ eW2 + b2
// ---------------------------------------------------------------------------

__global__ __launch_bounds__(256) void edge_block_kernel(
    const float* __restrict__ nodes, const float* __restrict__ edges,
    const float* __restrict__ glat,
    const v16bf* __restrict__ w1f, const float* __restrict__ b1,
    const v16bf* __restrict__ w2f, const float* __restrict__ b2,
    const float* __restrict__ rn_w,
    const int* __restrict__ receivers, const int* __restrict__ senders,
    const int* __restrict__ egidx,
    float* __restrict__ out_edges, float* adjacent, float* edge_agg) {
  __shared__ __bf16 hstage[8][16 * DDIM];
  const int lane = threadIdx.x & 31;
  const int wave = threadIdx.x >> 5;
  const int tile = blockIdx.x * 8 + wave;   // 16 edges per wave tile
  const int nn = lane & 15;
  const int kh = lane >> 4;

  const int ea = tile * 16 + nn;  // A-side gather row for this lane
  const float* segs[4] = {
      edges + (size_t)ea * DDIM,
      nodes + (size_t)senders[ea] * DDIM,
      nodes + (size_t)receivers[ea] * DDIM,
      glat + (size_t)egidx[ea] * DDIM,
  };

  v8f acc[8] = {};
  gemm_layer1<16>(segs, w1f, lane, acc);          // K = 512

  bias_relu_store(acc, b1, hstage[wave], lane);

  v8f acc2[8] = {};
  gemm_layer2(hstage[wave], w2f, lane, acc2);     // K = 128

  const float mval = logf(1.f + expf(rn_w[0]));   // softplus resnet mult
  #pragma unroll
  for (int ct = 0; ct < 8; ++ct) {
    const int c = ct * 16 + nn;
    const float bv = b2[c];
    #pragma unroll
    for (int r = 0; r < 8; ++r) {
      const int e = tile * 16 + (r + 8 * kh);
      const float u = acc2[ct][r] + bv;
      const size_t idx = (size_t)e * DDIM + c;
      out_edges[idx] = edges[idx] + mval * u;
      atomicAdd(adjacent + (size_t)receivers[e] * DDIM + c, u);
      atomicAdd(edge_agg + (size_t)egidx[e] * DDIM + c, u);
    }
  }
}

// ---------------------------------------------------------------------------
// Node block: u = relu([nodes|adjacent|g] @ nW1 + b1) @ nW2 + b2
// ---------------------------------------------------------------------------

__global__ __launch_bounds__(256) void node_block_kernel(
    const float* __restrict__ nodes, const float* __restrict__ adjacent,
    const float* __restrict__ glat,
    const v16bf* __restrict__ w1f, const float* __restrict__ b1,
    const v16bf* __restrict__ w2f, const float* __restrict__ b2,
    const float* __restrict__ rn_w,
    const int* __restrict__ ngidx,
    float* __restrict__ out_nodes, float* node_agg) {
  __shared__ __bf16 hstage[8][16 * DDIM];
  const int lane = threadIdx.x & 31;
  const int wave = threadIdx.x >> 5;
  const int tile = blockIdx.x * 8 + wave;   // 16 nodes per wave tile
  const int nn = lane & 15;
  const int kh = lane >> 4;

  const int va = tile * 16 + nn;
  const float* segs[4] = {
      nodes + (size_t)va * DDIM,
      adjacent + (size_t)va * DDIM,
      glat + (size_t)ngidx[va] * DDIM,
      glat + (size_t)ngidx[va] * DDIM,   // unused (KT=12)
  };

  v8f acc[8] = {};
  gemm_layer1<12>(segs, w1f, lane, acc);          // K = 384

  bias_relu_store(acc, b1, hstage[wave], lane);

  v8f acc2[8] = {};
  gemm_layer2(hstage[wave], w2f, lane, acc2);     // K = 128

  const float mval = logf(1.f + expf(rn_w[0]));
  #pragma unroll
  for (int ct = 0; ct < 8; ++ct) {
    const int c = ct * 16 + nn;
    const float bv = b2[c];
    #pragma unroll
    for (int r = 0; r < 8; ++r) {
      const int v = tile * 16 + (r + 8 * kh);
      const float u = acc2[ct][r] + bv;
      const size_t idx = (size_t)v * DDIM + c;
      out_nodes[idx] = nodes[idx] + mval * u;
      atomicAdd(node_agg + (size_t)ngidx[v] * DDIM + c, u);
    }
  }
}

// ---------------------------------------------------------------------------
// Global block: one wave, B = 16 rows exactly one WMMA M tile.
// ---------------------------------------------------------------------------

__global__ __launch_bounds__(32) void global_block_kernel(
    const float* __restrict__ glat,
    const float* __restrict__ node_agg, const float* __restrict__ edge_agg,
    const v16bf* __restrict__ w1f, const float* __restrict__ b1,
    const v16bf* __restrict__ w2f, const float* __restrict__ b2,
    const float* __restrict__ rn_w,
    float* __restrict__ out_glob) {
  __shared__ __bf16 hstage[16 * DDIM];
  const int lane = threadIdx.x & 31;
  const int nn = lane & 15;
  const int kh = lane >> 4;

  const float* segs[4] = {
      node_agg + (size_t)nn * DDIM,
      edge_agg + (size_t)nn * DDIM,
      glat + (size_t)nn * DDIM,
      glat + (size_t)nn * DDIM,   // unused (KT=12)
  };

  v8f acc[8] = {};
  gemm_layer1<12>(segs, w1f, lane, acc);          // K = 384

  bias_relu_store(acc, b1, hstage, lane);

  v8f acc2[8] = {};
  gemm_layer2(hstage, w2f, lane, acc2);           // K = 128

  const float mval = logf(1.f + expf(rn_w[0]));
  #pragma unroll
  for (int ct = 0; ct < 8; ++ct) {
    const int c = ct * 16 + nn;
    const float bv = b2[c];
    #pragma unroll
    for (int r = 0; r < 8; ++r) {
      const int b = r + 8 * kh;   // batch row
      const float u = acc2[ct][r] + bv;
      const size_t idx = (size_t)b * DDIM + c;
      out_glob[idx] = glat[idx] + mval * u;
    }
  }
}

// ---------------------------------------------------------------------------
// Launch
// ---------------------------------------------------------------------------

extern "C" void kernel_launch(void* const* d_in, const int* in_sizes, int n_in,
                              void* d_out, int out_size, void* d_ws,
                              size_t ws_size, hipStream_t stream) {
  (void)in_sizes; (void)n_in; (void)out_size; (void)ws_size;

  const float* nodes = (const float*)d_in[0];
  const float* edges = (const float*)d_in[1];
  const float* glat  = (const float*)d_in[2];
  const float* eW1 = (const float*)d_in[3];  const float* eb1 = (const float*)d_in[4];
  const float* eW2 = (const float*)d_in[5];  const float* eb2 = (const float*)d_in[6];
  const float* nW1 = (const float*)d_in[7];  const float* nb1 = (const float*)d_in[8];
  const float* nW2 = (const float*)d_in[9];  const float* nb2 = (const float*)d_in[10];
  const float* gW1 = (const float*)d_in[11]; const float* gb1 = (const float*)d_in[12];
  const float* gW2 = (const float*)d_in[13]; const float* gb2 = (const float*)d_in[14];
  const float* rn_w = (const float*)d_in[15];
  const int* receivers = (const int*)d_in[16];
  const int* senders   = (const int*)d_in[17];
  const int* ngidx     = (const int*)d_in[18];
  const int* egidx     = (const int*)d_in[19];

  float* out_nodes = (float*)d_out;
  float* out_edges = out_nodes + (size_t)NNODES * DDIM;
  float* out_glob  = out_edges + (size_t)NEDGES * DDIM;

  // Workspace layout (all offsets 32B-aligned for v16bf loads):
  //   adjacent [N,128] f32 | node_agg [16,128] | edge_agg [16,128] |
  //   packed bf16 weights: eW1p eW2p nW1p nW2p gW1p gW2p
  char* ws = (char*)d_ws;
  float* adjacent = (float*)ws;
  float* node_agg = adjacent + (size_t)NNODES * DDIM;
  float* edge_agg = node_agg + (size_t)NBATCH * DDIM;
  char* wbase = (char*)(edge_agg + (size_t)NBATCH * DDIM);
  __bf16* eW1p = (__bf16*)(wbase);
  __bf16* eW2p = (__bf16*)(wbase + 131072);
  __bf16* nW1p = (__bf16*)(wbase + 131072 + 32768);
  __bf16* nW2p = (__bf16*)(wbase + 131072 + 32768 + 98304);
  __bf16* gW1p = (__bf16*)(wbase + 131072 + 32768 + 98304 + 32768);
  __bf16* gW2p = (__bf16*)(wbase + 131072 + 32768 + 98304 + 32768 + 98304);

  // 1) zero the accumulation buffers (adjacent + node_agg + edge_agg, contiguous)
  const int zn = NNODES * DDIM + 2 * NBATCH * DDIM;
  zero_f32_kernel<<<4096, 256, 0, stream>>>(adjacent, zn);

  // 2) pack weights into bf16 B-fragment order
  pack_weights_kernel<<<(4 * DDIM * DDIM) / 256, 256, 0, stream>>>(eW1, eW1p, 4 * DDIM);
  pack_weights_kernel<<<(DDIM * DDIM) / 256, 256, 0, stream>>>(eW2, eW2p, DDIM);
  pack_weights_kernel<<<(3 * DDIM * DDIM) / 256, 256, 0, stream>>>(nW1, nW1p, 3 * DDIM);
  pack_weights_kernel<<<(DDIM * DDIM) / 256, 256, 0, stream>>>(nW2, nW2p, DDIM);
  pack_weights_kernel<<<(3 * DDIM * DDIM) / 256, 256, 0, stream>>>(gW1, gW1p, 3 * DDIM);
  pack_weights_kernel<<<(DDIM * DDIM) / 256, 256, 0, stream>>>(gW2, gW2p, DDIM);

  // 3) edge block (also scatters edges_update into adjacent / edge_agg)
  edge_block_kernel<<<NEDGES / (16 * 8), 256, 0, stream>>>(
      nodes, edges, glat, (const v16bf*)eW1p, eb1, (const v16bf*)eW2p, eb2,
      rn_w, receivers, senders, egidx, out_edges, adjacent, edge_agg);

  // 4) node block (reads adjacent, scatters nodes_update into node_agg)
  node_block_kernel<<<NNODES / (16 * 8), 256, 0, stream>>>(
      nodes, adjacent, glat, (const v16bf*)nW1p, nb1, (const v16bf*)nW2p, nb2,
      rn_w, ngidx, out_nodes, node_agg);

  // 5) global block (single wave, 16-row tile)
  global_block_kernel<<<1, 32, 0, stream>>>(
      glat, node_agg, edge_agg, (const v16bf*)gW1p, gb1, (const v16bf*)gW2p,
      gb2, rn_w, out_glob);
}
